// Dense_Attention_39273180954819
// MI455X (gfx1250) — matrix operand order repaired and checked by reading, compiled, not run on hardware
//
#include <hip/hip_runtime.h>

// ============================================================================
// Fused windowed attention (CrossFormer-style, dynamic pos bias) for MI455X
// gfx1250. One workgroup (4 wave32) per 8x8 window; all matmuls on
// v_wmma_f32_16x16x32_f16 (f16 in, f32 accumulate). Single fused pass over
// HBM (~112 MB => ~5us floor at 23.3 TB/s).
// Round 4: tensor_load_to_lds uses the 6-arg builtin form this toolchain
// declares (extra int32x8 group, zero-filled). x-window is DMA'd into LDS by
// the Tensor Data Mover (TENSORcnt + s_wait_tensorcnt), with edge
// zero-padding via the TDM tensor-dim OOB rule; qkv scatter-store uses a
// precomputed descriptor table; projection stores use constant-stride
// addressing.
// ============================================================================

typedef _Float16 v16h __attribute__((ext_vector_type(16)));
typedef _Float16 v8h  __attribute__((ext_vector_type(8)));
typedef float    v8f  __attribute__((ext_vector_type(8)));
typedef unsigned int u32x4 __attribute__((ext_vector_type(4)));
typedef int          i32x8 __attribute__((ext_vector_type(8)));
typedef int          i32x4 __attribute__((ext_vector_type(4)));

#define NEG_INF  (-1000000.0f)
#define Himg  250
#define Wimg  250
#define DIMC  28
#define NHEAD 4
#define HDIM  7
#define GG    64
#define NBLK  8192          // 8 images * 32*32 windows

// ---- LDS layout (byte offsets, all 16B aligned) ----
#define SX_OFF       0        // [64][32]  f16   4096
#define WQKV_OFF     4096     // [96][32]  f16   6144 (rows>=84, cols>=28 zero)
#define WPROJ_OFF    10240    // [32][32]  f16   2048 (rows/cols>=28 zero)
#define SQ_OFF       12288    // [4][64][8] f16  4096
#define SK_OFF       16384    // [4][64][8] f16  4096
#define SVT_OFF      20480    // [4][8][64] f16  4096 (transposed: [h][d][tok])
#define SPOST_OFF    24576    // [4][225]  f32   3600 (+pad)
#define SMASK_OFF    28192    // [64] f32         256
#define SZERO_OFF    28448    // [64] f16         128 (zero row for ptr-select)
#define SATTN_OFF    28576    // [4][64][64] f32 65536 (aliased: TDM f32 staging)
#define SATTN16_OFF  94112    // [4][64][64] f16 32768
#define SOUT_OFF     126880   // [64][28] f32    7168
#define SOUT16_OFF   134048   // [64][32] f16    4096
#define SQKVB_OFF    138144   // [84] f32         336 (+pad)
#define SPROJB_OFF   138496   // [28] f32         112 (+pad)
#define SDESC_OFF    138624   // [96] i32         384 (qkv store descriptors)
#define SMEM_TOTAL   139008   // ~136 KB -> 2 workgroups / 320KB WGP

__device__ __forceinline__ v8h ld8(const _Float16* p) {
  return *(const v8h*)p;
}
__device__ __forceinline__ v16h mk16(v8h lo, v8h hi) {
  return __builtin_shufflevector(lo, hi, 0,1,2,3,4,5,6,7,8,9,10,11,12,13,14,15);
}

// ----------------------------------------------------------------------------
// DynamicPosBias MLP over the 225 relative offsets (PDIM==1: LayerNorm over a
// size-1 axis computed faithfully). Writes pos[225][NHEAD] into workspace.
// ----------------------------------------------------------------------------
__global__ void pos_mlp_kernel(const float* __restrict__ pw,  const float* __restrict__ pb,
                               const float* __restrict__ l1g, const float* __restrict__ l1b,
                               const float* __restrict__ f1w, const float* __restrict__ f1b,
                               const float* __restrict__ l2g, const float* __restrict__ l2b,
                               const float* __restrict__ f2w, const float* __restrict__ f2b,
                               const float* __restrict__ l3g, const float* __restrict__ l3b,
                               const float* __restrict__ f3w, const float* __restrict__ f3b,
                               float* __restrict__ pos) {
  int i = blockIdx.x * blockDim.x + threadIdx.x;
  if (i >= 225) return;
  float b0 = (float)(i / 15) - 7.0f;
  float b1 = (float)(i % 15) - 7.0f;
  float t = b0 * pw[0] + b1 * pw[1] + pb[0];
  float m, var;
  m = t; var = 0.0f;
  t = (t - m) * rsqrtf(var + 1e-5f) * l1g[0] + l1b[0];
  t = fmaxf(t, 0.0f) * f1w[0] + f1b[0];
  m = t; var = 0.0f;
  t = (t - m) * rsqrtf(var + 1e-5f) * l2g[0] + l2b[0];
  t = fmaxf(t, 0.0f) * f2w[0] + f2b[0];
  m = t; var = 0.0f;
  t = (t - m) * rsqrtf(var + 1e-5f) * l3g[0] + l3b[0];
  float tr = fmaxf(t, 0.0f);
  #pragma unroll
  for (int h = 0; h < NHEAD; ++h) pos[i * NHEAD + h] = tr * f3w[h] + f3b[h];
}

// ----------------------------------------------------------------------------
// Main fused kernel. 128 threads = 4 wave32; wave = head in attention phases,
// wave = M-tile in projection phases.
//
// WMMA fragment packing (wave32, CDNA5 ISA 7.12.2), hf = lane>>4, lm = lane&15:
//   A (16x32 f16): lane m=lm; a[0..7]  <- K = hf*8 + 0..7
//                             a[8..15] <- K = 16 + hf*8 + 0..7
//   B (32x16 f16): lane n=lm; b[0..7] <- K = hf*8 + 0..7; b[8..15] <- K = 16 + hf*8 + 0..7
//   C (16x16 f32): lane n=lm; VGPR r -> M = hf*8 + r
// => every fragment = two 16B contiguous runs => ds_load_b128 pairs.
// ----------------------------------------------------------------------------
__global__ __launch_bounds__(128, 1)
void dense_attn_kernel(const float* __restrict__ x,
                       const float* __restrict__ qkv_w, const float* __restrict__ qkv_b,
                       const float* __restrict__ proj_w, const float* __restrict__ proj_b,
                       const float* __restrict__ pos, float* __restrict__ out) {
  extern __shared__ unsigned char smem_raw[];
  _Float16* sx     = (_Float16*)(smem_raw + SX_OFF);
  _Float16* wqkv   = (_Float16*)(smem_raw + WQKV_OFF);
  _Float16* wproj  = (_Float16*)(smem_raw + WPROJ_OFF);
  _Float16* sq     = (_Float16*)(smem_raw + SQ_OFF);
  _Float16* sk     = (_Float16*)(smem_raw + SK_OFF);
  _Float16* svT    = (_Float16*)(smem_raw + SVT_OFF);
  float*    sposT  = (float*)   (smem_raw + SPOST_OFF);
  float*    smask  = (float*)   (smem_raw + SMASK_OFF);
  _Float16* szero  = (_Float16*)(smem_raw + SZERO_OFF);
  float*    sattn  = (float*)   (smem_raw + SATTN_OFF);
  const float* staging = (const float*)(smem_raw + SATTN_OFF);  // TDM f32 tile
  _Float16* sattn16= (_Float16*)(smem_raw + SATTN16_OFF);
  float*    sout   = (float*)   (smem_raw + SOUT_OFF);
  _Float16* sout16 = (_Float16*)(smem_raw + SOUT16_OFF);
  float*    sqkvb  = (float*)   (smem_raw + SQKVB_OFF);
  float*    sprojb = (float*)   (smem_raw + SPROJB_OFF);
  int*      sdesc  = (int*)     (smem_raw + SDESC_OFF);

  const int tid  = threadIdx.x;
  const int lane = tid & 31;
  const int wave = tid >> 5;
  const int hf   = lane >> 4;
  const int lm   = lane & 15;

  const int blk  = blockIdx.x;
  const int bimg = blk >> 10;
  const int wloc = blk & 1023;
  const int wh   = wloc >> 5;
  const int ww   = wloc & 31;

  const v8h z8 = {};

  // ---------------- Phase 0a: issue TDM load of the x window ---------------
  // 2D tile: 8 rows of 8*28 contiguous f32, row stride 250*28 elements.
  // tensor_dim0/1 are the *clamped* valid extents so the TDM zero-fills the
  // out-of-image region of the tile for us (ISA 8.2: OOB reads return zero).
  if (wave == 0) {
    const unsigned td0 = (unsigned)((Wimg - ww * 8 < 8 ? Wimg - ww * 8 : 8) * DIMC);
    const unsigned td1 = (unsigned)(Himg - wh * 8 < 8 ? Himg - wh * 8 : 8);
    const unsigned long long ga = (unsigned long long)(const void*)
        (x + ((size_t)bimg * 62500 + (size_t)(wh * 8) * 250 + (size_t)(ww * 8)) * DIMC);
    u32x4 g0;
    g0[0] = 1u;                                        // count=1 valid descriptor
    g0[1] = (unsigned)SATTN_OFF;                       // lds_addr (staging)
    g0[2] = (unsigned)(ga & 0xFFFFFFFFu);              // global_addr[31:0]
    g0[3] = (unsigned)((ga >> 32) & 0x01FFFFFFu) | (2u << 30);  // addr[56:32]|type=2
    i32x8 g1;
    g1[0] = (int)(2u << 16);                           // data_size=2 (4 bytes)
    g1[1] = (int)((td0 & 0xFFFFu) << 16);              // tensor_dim0[15:0]
    g1[2] = (int)(((td0 >> 16) & 0xFFFFu) | ((td1 & 0xFFFFu) << 16));
    g1[3] = (int)(((td1 >> 16) & 0xFFFFu) | (224u << 16));  // tile_dim0 = 224
    g1[4] = 8;                                         // tile_dim1 = 8 rows
    g1[5] = Wimg * DIMC;                               // tensor_dim0_stride = 7000
    g1[6] = 0;
    g1[7] = 0;
    i32x4 gz4 = {};
    i32x8 gz8 = {};
    __builtin_amdgcn_tensor_load_to_lds(g0, g1, gz4, gz4, gz8, 0);
  }

  // ---------------- Phase 0b: stage weights / tables (overlaps TDM) --------
  for (int i = tid; i < 96 * 32; i += 128) {          // Wqkv, zero-padded
    int row = i >> 5, col = i & 31;
    wqkv[i] = (row < 84 && col < DIMC) ? (_Float16)qkv_w[row * DIMC + col] : (_Float16)0;
  }
  for (int i = tid; i < 32 * 32; i += 128) {          // Wproj, zero-padded
    int row = i >> 5, col = i & 31;
    wproj[i] = (row < DIMC && col < DIMC) ? (_Float16)proj_w[row * DIMC + col] : (_Float16)0;
  }
  for (int i = tid; i < NHEAD * 225; i += 128) {      // pos, transposed [h][225]
    int h = i / 225, j = i - h * 225;
    sposT[i] = pos[j * NHEAD + h];
  }
  for (int i = tid; i < 84; i += 128) sqkvb[i] = qkv_b[i];
  for (int i = tid; i < DIMC; i += 128) sprojb[i] = proj_b[i];
  for (int i = tid; i < 3 * NHEAD * GG * 8; i += 128) sq[i] = (_Float16)0; // sq/sk/svT contig
  for (int i = tid; i < GG; i += 128) szero[i] = (_Float16)0;
  for (int i = tid; i < 96; i += 128) {               // qkv store descriptors
    int desc = -1;
    if (i < 84) {
      int which = i / 28, ch = i % 28, h = ch / 7, d = ch % 7;
      int off, flags;
      if (which == 2)      { off = SVT_OFF + ((h * 8 + d) * GG) * 2;   flags = 1; } // v
      else if (which == 1) { off = SK_OFF  + (h * GG * 8 + d) * 2;     flags = 0; } // k
      else                 { off = SQ_OFF  + (h * GG * 8 + d) * 2;     flags = 2; } // q
      desc = (off << 2) | flags;
    }
    sdesc[i] = desc;
  }
  if (tid < GG) {
    int gr = tid >> 3, gc = tid & 7;
    smask[tid] = ((wh * 8 + gr) < Himg && (ww * 8 + gc) < Wimg) ? 0.0f : NEG_INF;
  }
  if (wave == 0) __builtin_amdgcn_s_wait_tensorcnt(0);
  __syncthreads();

  // ---------------- Phase 0c: staging f32 -> sx f16 [64][32] ---------------
  for (int i = tid; i < GG * 32; i += 128) {
    int g = i >> 5, ch = i & 31;
    int gr = g >> 3, gc = g & 7;
    sx[i] = (ch < DIMC) ? (_Float16)staging[gr * 224 + gc * DIMC + ch] : (_Float16)0;
  }
  __syncthreads();

  // ---------------- Phase 1: QKV = X (64x28) @ Wqkv^T, WMMA ----------------
  // wave owns M-tile = wave; A fragment reused across 6 N-tiles (96-wide out)
  {
    const _Float16* xr = sx + (wave * 16 + lm) * 32;
    const v16h a = mk16(ld8(xr + hf * 8), ld8(xr + 16 + hf * 8));
    #pragma unroll
    for (int nt = 0; nt < 6; ++nt) {
      const int ng = nt * 16 + lm;                       // < 96, rows>=84 are zero
      const _Float16* wr = wqkv + ng * 32 + hf * 8;
      const v16h bf = mk16(ld8(wr), ld8(wr + 16));
      v8f cz = {};
      v8f c = __builtin_amdgcn_wmma_f32_16x16x32_f16(false, a, false, bf, (short)0, cz, false, false);
      const int desc = sdesc[ng];
      if (desc >= 0) {
        const float bias = sqkvb[ng];
        _Float16* bp = (_Float16*)(smem_raw + (desc >> 2));
        if (desc & 1) {                                  // v: [h][d][tok], stride 1
          _Float16* dp = bp + wave * 16 + (hf << 3);
          #pragma unroll
          for (int r = 0; r < 8; ++r) dp[r] = (_Float16)(c[r] + bias);
        } else {                                         // q/k: [h][tok][8], stride 8
          const float scl = (desc & 2) ? 0.3779644730092272f : 1.0f; // HD^-0.5 in q
          _Float16* dp = bp + (wave * 16 + (hf << 3)) * 8;
          #pragma unroll
          for (int r = 0; r < 8; ++r) dp[r * 8] = (_Float16)((c[r] + bias) * scl);
        }
      }
    }
  }
  __syncthreads();

  // ---------------- Phase 2: scores = q @ k^T (64x64 per head), WMMA -------
  // wave = head; K=7 lives in halves 0..7 (hf==0 lanes); zero-row ptr-select
  // keeps loads unconditional and EXEC all-ones. Bias+mask fused into store.
  {
    const _Float16* qh = sq + wave * GG * 8;
    const _Float16* kh = sk + wave * GG * 8;
    float* ah = sattn + wave * GG * GG;
    const float* posh = sposT + wave * 225;
    v16h kb[4];
    #pragma unroll
    for (int nt = 0; nt < 4; ++nt) {
      const _Float16* kp = hf ? szero : (kh + (nt * 16 + lm) * 8);
      kb[nt] = mk16(ld8(kp), z8);
    }
    #pragma unroll
    for (int mt = 0; mt < 4; ++mt) {
      const _Float16* qp = hf ? szero : (qh + (mt * 16 + lm) * 8);
      const v16h a = mk16(ld8(qp), z8);
      #pragma unroll
      for (int nt = 0; nt < 4; ++nt) {
        v8f cz = {};
        v8f c = __builtin_amdgcn_wmma_f32_16x16x32_f16(false, a, false, kb[nt], (short)0, cz, false, false);
        const int tn = nt * 16 + lm;
        const float mask = smask[tn];
        // tm>>3 == 2*mt+hf is constant over r => bias index is base+r (stride-1)
        const int di = (2 * mt + hf) - (tn >> 3);
        const int base = (di + 7) * 15 + 7 - (tn & 7);
        float* arow = ah + (mt * 16 + (hf << 3)) * GG + tn;
        #pragma unroll
        for (int r = 0; r < 8; ++r)
          arow[r * GG] = c[r] + posh[base + r] + mask;
      }
    }
  }
  __syncthreads();

  // ---------------- Phase 3: row softmax (regs), requantize to f16 ---------
  {
    float* ah = sattn + wave * GG * GG;
    _Float16* ah16 = sattn16 + wave * GG * GG;
    for (int rr = 0; rr < 2; ++rr) {
      const int row = rr * 32 + lane;
      const float4* arow4 = (const float4*)(ah + row * GG);
      float e[GG];
      #pragma unroll
      for (int j4 = 0; j4 < 16; ++j4) {
        float4 t = arow4[j4];
        e[4 * j4 + 0] = t.x; e[4 * j4 + 1] = t.y;
        e[4 * j4 + 2] = t.z; e[4 * j4 + 3] = t.w;
      }
      float mx = -3.4e38f;
      #pragma unroll
      for (int j = 0; j < GG; ++j) mx = fmaxf(mx, e[j]);
      float s = 0.0f;
      #pragma unroll
      for (int j = 0; j < GG; ++j) { e[j] = __expf(e[j] - mx); s += e[j]; }
      const float inv = 1.0f / s;
      _Float16* orow = ah16 + row * GG;
      #pragma unroll
      for (int j = 0; j < GG; ++j) orow[j] = (_Float16)(e[j] * inv);
    }
  }
  __syncthreads();

  // ---------------- Phase 4: out_h = attn (64x64) @ v (64x7), WMMA ---------
  // B fragments (from transposed v) depend only on ko: 2 loads, reused 4x.
  {
    const _Float16* ah16 = sattn16 + wave * GG * GG;
    v16h vb[2];
    #pragma unroll
    for (int koi = 0; koi < 2; ++koi) {
      const _Float16* vp = (lm < 8)
          ? (svT + (wave * 8 + lm) * GG + koi * 32 + hf * 8)
          : szero;
      vb[koi] = mk16(ld8(vp), ld8(vp + 16));
    }
    #pragma unroll
    for (int mt = 0; mt < 4; ++mt) {
      v8f acc = {};
      #pragma unroll
      for (int koi = 0; koi < 2; ++koi) {
        const _Float16* ar = ah16 + (mt * 16 + lm) * GG + koi * 32;
        const v16h a = mk16(ld8(ar + hf * 8), ld8(ar + 16 + hf * 8));
        acc = __builtin_amdgcn_wmma_f32_16x16x32_f16(false, a, false, vb[koi], (short)0, acc, false, false);
      }
      if (lm < HDIM) {
        float* op = sout + (mt * 16 + (hf << 3)) * DIMC + wave * HDIM + lm;
        #pragma unroll
        for (int r = 0; r < 8; ++r) op[r * DIMC] = acc[r];
      }
    }
  }
  __syncthreads();

  // ---------------- Phase 5: pack concat(heads) to f16 ---------------------
  for (int i = tid; i < GG * 32; i += 128) {
    int g = i >> 5, nn = i & 31;
    sout16[i] = (nn < DIMC) ? (_Float16)sout[g * DIMC + nn] : (_Float16)0;
  }
  __syncthreads();

  // ---------------- Phase 6: proj (64x28)@Wproj^T, fused crop + store ------
  {
    const _Float16* orow = sout16 + (wave * 16 + lm) * 32;
    const v16h a = mk16(ld8(orow + hf * 8), ld8(orow + 16 + hf * 8));
    const int cmax = Wimg - ww * 8;                    // uniform column clamp
    #pragma unroll
    for (int nt = 0; nt < 2; ++nt) {
      const int ng = nt * 16 + lm;                     // < 32, rows>=28 zero
      const _Float16* wr = wproj + ng * 32 + hf * 8;
      const v16h bf = mk16(ld8(wr), ld8(wr + 16));
      v8f cz = {};
      v8f c = __builtin_amdgcn_wmma_f32_16x16x32_f16(false, a, false, bf, (short)0, cz, false, false);
      if (ng < DIMC) {
        // tok>>3 == 2*wave+hf is constant per lane: one row base, stride 112B
        const int rr2 = wh * 8 + wave * 2 + hf;
        if (rr2 < Himg) {
          const float pbv = sprojb[ng];
          float* op = out + ((size_t)bimg * 62500 + (size_t)rr2 * 250 + (size_t)(ww * 8)) * DIMC + ng;
          #pragma unroll
          for (int r = 0; r < 8; ++r)
            if (r < cmax) op[r * DIMC] = c[r] + pbv;
        }
      }
    }
  }
}

// ----------------------------------------------------------------------------
extern "C" void kernel_launch(void* const* d_in, const int* in_sizes, int n_in,
                              void* d_out, int out_size, void* d_ws, size_t ws_size,
                              hipStream_t stream) {
  const float* x      = (const float*)d_in[0];
  const float* qkv_w  = (const float*)d_in[1];
  const float* qkv_b  = (const float*)d_in[2];
  const float* proj_w = (const float*)d_in[3];
  const float* proj_b = (const float*)d_in[4];
  const float* ppw    = (const float*)d_in[5];
  const float* ppb    = (const float*)d_in[6];
  const float* l1g    = (const float*)d_in[7];
  const float* l1b    = (const float*)d_in[8];
  const float* f1w    = (const float*)d_in[9];
  const float* f1b    = (const float*)d_in[10];
  const float* l2g    = (const float*)d_in[11];
  const float* l2b    = (const float*)d_in[12];
  const float* f2w    = (const float*)d_in[13];
  const float* f2b    = (const float*)d_in[14];
  const float* l3g    = (const float*)d_in[15];
  const float* l3b    = (const float*)d_in[16];
  const float* f3w    = (const float*)d_in[17];
  const float* f3b    = (const float*)d_in[18];
  float* pos = (float*)d_ws;   // 225*4 floats of workspace

  (void)in_sizes; (void)n_in; (void)out_size; (void)ws_size;

  (void)hipFuncSetAttribute((const void*)dense_attn_kernel,
                            hipFuncAttributeMaxDynamicSharedMemorySize, SMEM_TOTAL);

  pos_mlp_kernel<<<1, 256, 0, stream>>>(ppw, ppb, l1g, l1b, f1w, f1b,
                                        l2g, l2b, f2w, f2b, l3g, l3b,
                                        f3w, f3b, pos);
  dense_attn_kernel<<<NBLK, 128, SMEM_TOTAL, stream>>>(x, qkv_w, qkv_b,
                                                       proj_w, proj_b, pos,
                                                       (float*)d_out);
}
